// LSTM_82111184765557
// MI455X (gfx1250) — compile-verified, matrix operand order checked by reading, not run on hardware
//
#include <hip/hip_runtime.h>
#include <hip/hip_bf16.h>
#include <stdint.h>

#define BSZ 256
#define TT  128
#define FF  5
#define HH  1024
#define NG  4096   // 4*H

typedef __attribute__((ext_vector_type(16))) __bf16 v16bf;
typedef __attribute__((ext_vector_type(8)))  float  v8f;

__device__ __forceinline__ unsigned short f2bf(float f) {
    unsigned int u = __float_as_uint(f);
    unsigned int r = u + 0x7FFFu + ((u >> 16) & 1u);   // round-to-nearest-even
    return (unsigned short)(r >> 16);
}
__device__ __forceinline__ float bf2f(unsigned short s) {
    return __uint_as_float(((unsigned int)s) << 16);
}
__device__ __forceinline__ float sigmoidf(float z) {
    return 1.0f / (1.0f + __expf(-z));
}

// U [K=1024, N=4096] fp32 row-major  ->  Ut [N=4096, K=1024] bf16 row-major
__global__ void lstm_cvt_u(const float* __restrict__ U, unsigned short* __restrict__ Ut) {
    int idx = blockIdx.x * 256 + threadIdx.x;      // idx = n*1024 + k (coalesced write)
    int n = idx >> 10;
    int k = idx & 1023;
    Ut[idx] = f2bf(U[(size_t)k * NG + n]);
}

__global__ void lstm_init(unsigned short* __restrict__ h0, float* __restrict__ c) {
    int idx = blockIdx.x * 256 + threadIdx.x;      // B*H = 262144 elements
    h0[idx] = 0;
    c[idx]  = 0.0f;
}

// One time step, fully fused:  z = h_prev @ U (+ x_t@W + b) ; gates ; c,h update.
// Grid: 64 blocks x 256 threads = 512 waves.
// Wave tile = 32(batch) x 16(hidden unit) x 4 gates = 8 independent WMMA chains;
// each weight (B) fragment is reused by two M-tiles -> 1.5 b128 loads per WMMA.
__global__ void lstm_step(const unsigned short* __restrict__ hprev,
                          unsigned short* __restrict__ hnext,
                          float* __restrict__ c,
                          const unsigned short* __restrict__ Ut,
                          const float* __restrict__ x,
                          const float* __restrict__ W,
                          const float* __restrict__ bias,
                          int t) {
    const int wave = threadIdx.x >> 5;
    const int lane = threadIdx.x & 31;
    const int tile = blockIdx.x * 8 + wave;        // 0..511
    const int m0 = (tile >> 6) << 5;               // batch tile base (8 tiles of 32)
    const int n0 = (tile & 63) << 4;               // hidden-unit tile (64 tiles of 16)
    const int hl  = lane >> 4;                     // lane half (K phase select)
    const int col = lane & 15;                     // M for A-frag, N for B/C-frag

    v8f acc[2][4] = {{v8f{}, v8f{}, v8f{}, v8f{}},
                     {v8f{}, v8f{}, v8f{}, v8f{}}};

    const unsigned short* rowA0 = hprev + (size_t)(m0 + col) * HH;
    const unsigned short* rowA1 = rowA0 + (size_t)16 * HH;
    const unsigned short* rowB[4];
#pragma unroll
    for (int g = 0; g < 4; ++g)
        rowB[g] = Ut + (size_t)(g * HH + n0 + col) * HH;

    for (int kk = 0; kk < HH; kk += 32) {
        const int o0 = kk + hl * 8;                // 16B-aligned chunk, K = o0..o0+7
        const int o1 = kk + 16 + hl * 8;           // K = o1..o1+7
        v16bf a0, a1;
        {
            uint4* p = reinterpret_cast<uint4*>(&a0);
            p[0] = *reinterpret_cast<const uint4*>(rowA0 + o0);
            p[1] = *reinterpret_cast<const uint4*>(rowA0 + o1);
            uint4* q = reinterpret_cast<uint4*>(&a1);
            q[0] = *reinterpret_cast<const uint4*>(rowA1 + o0);
            q[1] = *reinterpret_cast<const uint4*>(rowA1 + o1);
        }
#pragma unroll
        for (int g = 0; g < 4; ++g) {
            v16bf b;
            uint4* p = reinterpret_cast<uint4*>(&b);
            p[0] = *reinterpret_cast<const uint4*>(rowB[g] + o0);
            p[1] = *reinterpret_cast<const uint4*>(rowB[g] + o1);
            acc[0][g] = __builtin_amdgcn_wmma_f32_16x16x32_bf16(
                false, a0, false, b, (short)0, acc[0][g], false, false);
            acc[1][g] = __builtin_amdgcn_wmma_f32_16x16x32_bf16(
                false, a1, false, b, (short)0, acc[1][g], false, false);
        }
    }

    // Epilogue: add x_t@W + bias, apply gates, update c and h.
    const int n = n0 + col;
    float Wc[4][FF];
    float bz[4];
#pragma unroll
    for (int g = 0; g < 4; ++g) {
        bz[g] = bias[g * HH + n];
#pragma unroll
        for (int f = 0; f < FF; ++f)
            Wc[g][f] = W[(size_t)f * NG + g * HH + n];
    }

#pragma unroll
    for (int s = 0; s < 2; ++s) {
#pragma unroll
        for (int r = 0; r < 8; ++r) {
            const int m = m0 + s * 16 + r + hl * 8;   // C/D layout: VGPR r -> M=r (+8 hi lanes)
            const float* xr = x + ((size_t)m * TT + t) * FF;
            float xv0 = xr[0], xv1 = xr[1], xv2 = xr[2], xv3 = xr[3], xv4 = xr[4];

            float z[4];
#pragma unroll
            for (int g = 0; g < 4; ++g) {
                float sum = acc[s][g][r] + bz[g];
                sum += xv0 * Wc[g][0] + xv1 * Wc[g][1] + xv2 * Wc[g][2]
                     + xv3 * Wc[g][3] + xv4 * Wc[g][4];
                z[g] = sum;
            }
            float ig = sigmoidf(z[0]);
            float fg = sigmoidf(z[1]);
            float gg = fmaxf(z[2], 0.0f);          // activation=relu on candidate
            float og = sigmoidf(z[3]);

            const size_t ci = (size_t)m * HH + n;
            float cn = fg * c[ci] + ig * gg;
            c[ci] = cn;
            float hn = og * fmaxf(cn, 0.0f);       // activation=relu on cell output
            hnext[ci] = f2bf(hn);
        }
    }
}

// out[b, j] = h_last[b,:] @ Wo[:, j] + bo[j]   (256 x 5, tiny)
__global__ void lstm_out(const unsigned short* __restrict__ h,
                         const float* __restrict__ Wo,
                         const float* __restrict__ bo,
                         float* __restrict__ out) {
    int tid = blockIdx.x * 256 + threadIdx.x;
    if (tid >= BSZ * 5) return;
    int b = tid / 5;
    int j = tid % 5;
    float s = bo[j];
    const unsigned short* hr = h + (size_t)b * HH;
    for (int k = 0; k < HH; ++k)
        s += bf2f(hr[k]) * Wo[(size_t)k * 5 + j];
    out[tid] = s;
}

extern "C" void kernel_launch(void* const* d_in, const int* in_sizes, int n_in,
                              void* d_out, int out_size, void* d_ws, size_t ws_size,
                              hipStream_t stream) {
    const float* x    = (const float*)d_in[0];   // [256,128,5]
    const float* W    = (const float*)d_in[1];   // [5,4096]
    const float* U    = (const float*)d_in[2];   // [1024,4096]
    const float* bias = (const float*)d_in[3];   // [4096]
    const float* Wo   = (const float*)d_in[4];   // [1024,5]
    const float* bo   = (const float*)d_in[5];   // [5]
    float* out        = (float*)d_out;           // [256,5]

    uint8_t* ws = (uint8_t*)d_ws;
    // Workspace layout (~10 MB total):
    unsigned short* Ut = (unsigned short*)(ws);                 // 4096*1024*2 = 8 MB
    unsigned short* h0 = (unsigned short*)(ws + 8388608);       // 512 KB
    unsigned short* h1 = (unsigned short*)(ws + 8912896);       // 512 KB
    float*          c  = (float*)(ws + 9437184);                // 1 MB

    // One-time (per call) weight convert + transpose to bf16 [N,K], and state init.
    lstm_cvt_u<<<(NG * HH) / 256, 256, 0, stream>>>(U, Ut);
    lstm_init<<<(BSZ * HH) / 256, 256, 0, stream>>>(h0, c);

    unsigned short* bufs[2] = {h0, h1};
    for (int t = 0; t < TT; ++t) {
        lstm_step<<<64, 256, 0, stream>>>(bufs[t & 1], bufs[(t + 1) & 1],
                                          c, Ut, x, W, bias, t);
    }
    // t=127 wrote bufs[0] -> final hidden state.
    lstm_out<<<(BSZ * 5 + 255) / 256, 256, 0, stream>>>(bufs[0], Wo, bo, out);
}
